// ArgumentGCN_34_9680856285903
// MI455X (gfx1250) — matrix operand order repaired and checked by reading, not compiled, not run on hardware
//
#include <hip/hip_runtime.h>
#include <hip/hip_bf16.h>
#include <math.h>

// Problem constants (reference: B, N, D = 32, 512, 256; STEPS = 2)
#define GB 32
#define GN 512
#define GD 256
#define GSTEPS 2

typedef __attribute__((ext_vector_type(16))) __bf16 v16bf;
typedef __attribute__((ext_vector_type(8)))  float  v8f;

__device__ __forceinline__ v8f wmma_bf16(v16bf a, v16bf b, v8f c) {
    // emits v_wmma_f32_16x16x32_bf16
    return __builtin_amdgcn_wmma_f32_16x16x32_bf16(
        /*neg_a=*/false, a, /*neg_b=*/false, b,
        /*c_mod=*/(short)0, c, /*reuse_a=*/false, /*reuse_b=*/false);
}

// Load 16 bf16 as two aligned 16B chunks (8 + 8 elements) into a WMMA operand.
__device__ __forceinline__ v16bf load_frag(const __bf16* base, int off0, int off1) {
    union { uint4 q[2]; v16bf v; } u;
    u.q[0] = *(const uint4*)(base + off0);
    u.q[1] = *(const uint4*)(base + off1);
    return u.v;
}

__device__ __forceinline__ float wave_reduce_add(float v) {
    #pragma unroll
    for (int off = 16; off > 0; off >>= 1) v += __shfl_xor(v, off, 32);
    return v;
}

// ---------------------------------------------------------------------------
// f32 -> bf16 bulk conversion (one float4 -> uint2 per thread).
// ---------------------------------------------------------------------------
__global__ void cvt_bf16_kernel(const float* __restrict__ src, __bf16* __restrict__ dst,
                                int n4) {
    int idx = blockIdx.x * 256 + threadIdx.x;
    if (idx < n4) {
        float4 x = ((const float4*)src)[idx];
        union { __bf16 h[4]; uint2 u; } c;
        c.h[0] = (__bf16)x.x; c.h[1] = (__bf16)x.y;
        c.h[2] = (__bf16)x.z; c.h[3] = (__bf16)x.w;
        ((uint2*)dst)[idx] = c.u;
    }
}

// ---------------------------------------------------------------------------
// Step-invariant prep: per-row reciprocal neighbor count AND binarized bf16
// adjacency (dd mask folded in): abin[src][b,i,j] = mask_i*mask_j*(i!=j)*(adj>0).
// One wave per (b,i) row.
// ---------------------------------------------------------------------------
__global__ void nn_bin_kernel(const float* __restrict__ a0, const float* __restrict__ a1,
                              const float* __restrict__ a2, const float* __restrict__ a3,
                              const int* __restrict__ mask, float* __restrict__ invnn,
                              __bf16* __restrict__ ab0, __bf16* __restrict__ ab1,
                              __bf16* __restrict__ ab2, __bf16* __restrict__ ab3) {
    int tid = threadIdx.x, lane = tid & 31, w = tid >> 5;
    int row = blockIdx.x * 8 + w;                 // 0 .. B*N-1
    int b = row / GN, i = row % GN;
    int mi = mask[row];
    size_t base = ((size_t)b * GN + i) * GN;

    const float* As[4] = {a0, a1, a2, a3};
    __bf16*      Bs[4] = {ab0, ab1, ab2, ab3};

    const __bf16 one = (__bf16)1.0f, zero = (__bf16)0.0f;
    float sum = 0.0f;
    for (int j = lane; j < GN; j += 32) {
        bool dd = (mi != 0) && (mask[b * GN + j] != 0) && (j != i);
        #pragma unroll
        for (int src = 0; src < 4; ++src) {
            bool on = dd && (As[src][base + j] > 0.0f);
            Bs[src][base + j] = on ? one : zero;
            sum += on ? 1.0f : 0.0f;
        }
    }
    sum = wave_reduce_add(sum);
    if (lane == 0) invnn[row] = 1.0f / fmaxf(sum, 1.0f);
}

// ---------------------------------------------------------------------------
// d_w = sigmoid(node . w_nw + b_nw). One wave per row (f32 inputs).
// ---------------------------------------------------------------------------
__global__ void dw_kernel(const float* __restrict__ nodeIn, const float* __restrict__ w_nw,
                          const float* __restrict__ b_nw, float* __restrict__ dwOut,
                          float* __restrict__ allwOut, int step) {
    int tid = threadIdx.x, lane = tid & 31, w = tid >> 5;
    int row = blockIdx.x * 8 + w;                 // 0 .. B*N-1
    const float* r = nodeIn + (size_t)row * GD;
    const float4* rv = (const float4*)(r + lane * 8);
    const float4* wv = (const float4*)(w_nw + lane * 8);
    float4 x0 = rv[0], x1 = rv[1], y0 = wv[0], y1 = wv[1];
    float sum = x0.x * y0.x + x0.y * y0.y + x0.z * y0.z + x0.w * y0.w
              + x1.x * y1.x + x1.y * y1.y + x1.z * y1.z + x1.w * y1.w;
    sum = wave_reduce_add(sum);
    if (lane == 0) {
        float v = 1.0f / (1.0f + expf(-(sum + b_nw[0])));
        dwOut[row] = v;
        int b = row / GN, i = row % GN;
        allwOut[(b * GSTEPS + step) * GN + i] = v;
    }
}

// ---------------------------------------------------------------------------
// Self projection GEMM (bf16 in, f32 out + bias):
//   selfInfo[b,i,d] = sum_k node[b,i,k] * w_self[d,k] + b_self[d]
// Tile 64(i) x 256(d); 8 waves = 4(M) x 2(N); 8 WMMA accs per wave.
// A fragments: K-interleaved (0-7 / 16-23 pattern); B fragments: same K-interleaved order.
// ---------------------------------------------------------------------------
__global__ void self_proj_kernel(const __bf16* __restrict__ nodeBf,
                                 const __bf16* __restrict__ wSelfBf,
                                 const float* __restrict__ b_self,
                                 float* __restrict__ selfInfo) {
    const int mt = blockIdx.x;                    // 0 .. B*(N/64)-1
    const int b  = mt / (GN / 64);
    const int m0 = (mt % (GN / 64)) * 64;
    const int tid = threadIdx.x, lane = tid & 31, w = tid >> 5;
    const int msub = w & 3, nhalf = w >> 2;
    const int n = lane & 15;
    const int koffA = (lane < 16) ? 0 : 8;
    const int koffB = (lane < 16) ? 0 : 16;
    const int i = m0 + msub * 16 + n;
    const __bf16* Arow = nodeBf + ((size_t)b * GN + i) * GD;

    v8f acc[8];
    #pragma unroll
    for (int t = 0; t < 8; ++t) acc[t] = (v8f){0.f,0.f,0.f,0.f,0.f,0.f,0.f,0.f};

    for (int k0 = 0; k0 < GD; k0 += 32) {
        v16bf a = load_frag(Arow, k0 + koffA, k0 + 16 + koffA);
        #pragma unroll
        for (int t = 0; t < 8; ++t) {
            int d = nhalf * 128 + t * 16 + n;
            v16bf bb = load_frag(wSelfBf + (size_t)d * GD, k0 + koffA, k0 + 16 + koffA);
            acc[t] = wmma_bf16(a, bb, acc[t]);
        }
    }

    const int hi8 = (lane < 16) ? 0 : 8;
    #pragma unroll
    for (int t = 0; t < 8; ++t) {
        int d = nhalf * 128 + t * 16 + n;
        float bias = b_self[d];
        #pragma unroll
        for (int v = 0; v < 8; ++v) {
            int irow = m0 + msub * 16 + v + hi8;
            size_t R = (size_t)b * GN + irow;
            selfInfo[R * GD + d] = acc[t][v] + bias;
        }
    }
}

// ---------------------------------------------------------------------------
// Transposed propagate projections (bf16 in, bf16 transposed out):
//   infoT[b,d,j] = (sum_k w[d,k] * node[b,j,k]) * (mask[b,j] ? d_w[b,j] : 0)
// A = weight rows (M = d), B = node rows (N = j): C is d-major so the per-
// column scale is lane-uniform and stores are 16-lane contiguous.
// blockIdx.y = src weight 0..3. Tile 64(d) x 256(j) per block.
// ---------------------------------------------------------------------------
__global__ void projT_kernel(const __bf16* __restrict__ nodeBf,
                             const __bf16* __restrict__ wBf,   // 4 x [D,D], contiguous
                             const float* __restrict__ dw, const int* __restrict__ mask,
                             __bf16* __restrict__ qAll) {      // 4 x [B,D,N], contiguous
    const int src = blockIdx.y;
    const __bf16* W = wBf + (size_t)src * GD * GD;
    __bf16*       Q = qAll + (size_t)src * GB * GD * GN;

    const int idx = blockIdx.x;                   // 0 .. B*8-1
    const int b   = idx >> 3;
    const int rem = idx & 7;
    const int d0  = (rem & 3) * 64;               // d tile (4 x 64 = 256)
    const int j0b = (rem >> 2) * 256;             // j half (2 x 256 = 512)
    const int tid = threadIdx.x, lane = tid & 31, w = tid >> 5;
    const int msub = w & 3, nhalf = w >> 2;
    const int n = lane & 15;
    const int koffA = (lane < 16) ? 0 : 8;
    const int koffB = (lane < 16) ? 0 : 16;
    const int dm = d0 + msub * 16 + n;
    const __bf16* Arow = W + (size_t)dm * GD;

    v8f acc[8];
    #pragma unroll
    for (int t = 0; t < 8; ++t) acc[t] = (v8f){0.f,0.f,0.f,0.f,0.f,0.f,0.f,0.f};

    for (int k0 = 0; k0 < GD; k0 += 32) {
        v16bf a = load_frag(Arow, k0 + koffA, k0 + 16 + koffA);
        #pragma unroll
        for (int t = 0; t < 8; ++t) {
            int j = j0b + nhalf * 128 + t * 16 + n;
            v16bf bb = load_frag(nodeBf + ((size_t)b * GN + j) * GD,
                                 k0 + koffA, k0 + 16 + koffA);
            acc[t] = wmma_bf16(a, bb, acc[t]);
        }
    }

    const int hi8 = (lane < 16) ? 0 : 8;
    #pragma unroll
    for (int t = 0; t < 8; ++t) {
        int j = j0b + nhalf * 128 + t * 16 + n;   // column j: fixed per lane
        size_t Rj = (size_t)b * GN + j;
        float scale = (mask[Rj] != 0) ? dw[Rj] : 0.0f;
        #pragma unroll
        for (int v = 0; v < 8; ++v) {
            int d = d0 + msub * 16 + v + hi8;
            Q[((size_t)b * GD + d) * GN + j] = (__bf16)(acc[t][v] * scale);
        }
    }
}

// ---------------------------------------------------------------------------
// Aggregation GEMM + epilogue (no LDS, no barriers, pure bf16 b128 loads).
//   acc[b,i,d] = sum_{src,j} abin_src[b,i,j] * infoT_src[b,d,j]
//   out = relu(self_info + acc * invnn)   (+ optional fused bf16 copy of out)
// ---------------------------------------------------------------------------
__global__ void agg_kernel(const __bf16* __restrict__ abAll,  // 4 x [B,N,N]
                           const __bf16* __restrict__ qAll,   // 4 x [B,D,N]
                           const float* __restrict__ selfInfo, const float* __restrict__ invnn,
                           float* __restrict__ out, __bf16* __restrict__ outBf) {
    const int mt = blockIdx.x;
    const int b  = mt / (GN / 64);
    const int m0 = (mt % (GN / 64)) * 64;
    const int tid = threadIdx.x, lane = tid & 31, w = tid >> 5;
    const int msub = w & 3, nhalf = w >> 2;
    const int n = lane & 15;
    const int koffA = (lane < 16) ? 0 : 8;
    const int koffB = (lane < 16) ? 0 : 16;
    const int i = m0 + msub * 16 + n;

    v8f acc[8];
    #pragma unroll
    for (int t = 0; t < 8; ++t) acc[t] = (v8f){0.f,0.f,0.f,0.f,0.f,0.f,0.f,0.f};

    for (int src = 0; src < 4; ++src) {
        const __bf16* Ar = abAll + (size_t)src * GB * GN * GN + ((size_t)b * GN + i) * GN;
        const __bf16* Qb = qAll  + (size_t)src * GB * GD * GN + (size_t)b * GD * GN;

        for (int j0 = 0; j0 < GN; j0 += 32) {
            v16bf a = load_frag(Ar, j0 + koffA, j0 + 16 + koffA);
            #pragma unroll
            for (int t = 0; t < 8; ++t) {
                int d = nhalf * 128 + t * 16 + n;
                v16bf bb = load_frag(Qb + (size_t)d * GN, j0 + koffA, j0 + 16 + koffA);
                acc[t] = wmma_bf16(a, bb, acc[t]);
            }
        }
    }

    const int hi8 = (lane < 16) ? 0 : 8;
    #pragma unroll
    for (int t = 0; t < 8; ++t) {
        int d = nhalf * 128 + t * 16 + n;
        #pragma unroll
        for (int v = 0; v < 8; ++v) {
            int irow = m0 + msub * 16 + v + hi8;
            size_t R = (size_t)b * GN + irow;
            float val = selfInfo[R * GD + d] + acc[t][v] * invnn[R];
            val = fmaxf(val, 0.0f);
            out[R * GD + d] = val;
            if (outBf) outBf[R * GD + d] = (__bf16)val;   // fused bf16 copy for next step
        }
    }
}

// ---------------------------------------------------------------------------
extern "C" void kernel_launch(void* const* d_in, const int* in_sizes, int n_in,
                              void* d_out, int out_size, void* d_ws, size_t ws_size,
                              hipStream_t stream) {
    (void)in_sizes; (void)n_in; (void)out_size; (void)ws_size;

    const float* node      = (const float*)d_in[0];
    const int*   node_mask = (const int*)  d_in[1];
    const float* adjA      = (const float*)d_in[2];
    const float* adjAre    = (const float*)d_in[3];
    const float* adjNo     = (const float*)d_in[4];
    const float* adjNore   = (const float*)d_in[5];
    const float* w_nw      = (const float*)d_in[6];
    const float* b_nw      = (const float*)d_in[7];
    const float* w_self    = (const float*)d_in[8];
    const float* b_self    = (const float*)d_in[9];
    const float* w_arg     = (const float*)d_in[10];
    const float* w_argp    = (const float*)d_in[11];
    const float* w_noe     = (const float*)d_in[12];
    const float* w_noep    = (const float*)d_in[13];

    float* out_node = (float*)d_out;                         // [B,N,D]
    float* out_allw = out_node + (size_t)GB * GN * GD;       // [B,STEPS,N]

    // Workspace carve-up (~143 MB total)
    char*  ws  = (char*)d_ws;
    size_t off = 0;
    auto alloc = [&](size_t bytes) -> void* {
        void* p = ws + off;
        off += (bytes + 255) & ~(size_t)255;
        return p;
    };
    float*  dw       = (float*) alloc((size_t)GB * GN * 4);
    float*  invnn    = (float*) alloc((size_t)GB * GN * 4);
    float*  nodeA    = (float*) alloc((size_t)GB * GN * GD * 4);     // step-1 f32 output
    float*  selfInfo = (float*) alloc((size_t)GB * GN * GD * 4);
    __bf16* nodeBf   = (__bf16*)alloc((size_t)GB * GN * GD * 2);     // current node, bf16
    __bf16* wSelfBf  = (__bf16*)alloc((size_t)GD * GD * 2);
    __bf16* wPropBf  = (__bf16*)alloc((size_t)4 * GD * GD * 2);      // 4 weights contiguous
    __bf16* qAll     = (__bf16*)alloc((size_t)4 * GB * GD * GN * 2); // infoT [src,b,d,j]
    __bf16* abAll    = (__bf16*)alloc((size_t)4 * GB * GN * GN * 2); // binary adj bf16

    const int rowBlocks  = GB * GN / 8;            // 2048 (wave-per-row kernels)
    const int tileBlocks = GB * (GN / 64);         // 256  (GEMM tiles)
    const int wq4        = GD * GD / 4;            // float4s per weight matrix

    // --- once: weight conversion + fused degree/binarize ---
    cvt_bf16_kernel<<<(wq4 + 255) / 256, 256, 0, stream>>>(w_self, wSelfBf, wq4);
    cvt_bf16_kernel<<<(wq4 + 255) / 256, 256, 0, stream>>>(w_arg,  wPropBf + 0 * GD * GD, wq4);
    cvt_bf16_kernel<<<(wq4 + 255) / 256, 256, 0, stream>>>(w_argp, wPropBf + 1 * GD * GD, wq4);
    cvt_bf16_kernel<<<(wq4 + 255) / 256, 256, 0, stream>>>(w_noe,  wPropBf + 2 * GD * GD, wq4);
    cvt_bf16_kernel<<<(wq4 + 255) / 256, 256, 0, stream>>>(w_noep, wPropBf + 3 * GD * GD, wq4);

    nn_bin_kernel<<<rowBlocks, 256, 0, stream>>>(
        adjA, adjAre, adjNo, adjNore, node_mask, invnn,
        abAll + (size_t)0 * GB * GN * GN, abAll + (size_t)1 * GB * GN * GN,
        abAll + (size_t)2 * GB * GN * GN, abAll + (size_t)3 * GB * GN * GN);

    // step-1 node -> bf16 (subsequent steps: fused into agg epilogue)
    const int nq4 = GB * GN * GD / 4;
    cvt_bf16_kernel<<<(nq4 + 255) / 256, 256, 0, stream>>>(node, nodeBf, nq4);

    for (int s = 0; s < GSTEPS; ++s) {
        const float* nin   = (s == 0) ? node : nodeA;
        float*       nout  = (s == GSTEPS - 1) ? out_node : nodeA;
        __bf16*      nbfOut = (s == GSTEPS - 1) ? (__bf16*)nullptr : nodeBf;

        dw_kernel<<<rowBlocks, 256, 0, stream>>>(nin, w_nw, b_nw, dw, out_allw, s);

        self_proj_kernel<<<tileBlocks, 256, 0, stream>>>(nodeBf, wSelfBf, b_self, selfInfo);

        projT_kernel<<<dim3(tileBlocks, 4), 256, 0, stream>>>(
            nodeBf, wPropBf, dw, node_mask, qAll);

        agg_kernel<<<tileBlocks, 256, 0, stream>>>(
            abAll, qAll, selfInfo, invnn, nout, nbfOut);
    }
}
